// CNN_MAMBA2_14018773254553
// MI455X (gfx1250) — compile-verified, matrix operand order checked by reading, not run on hardware
//
#include <hip/hip_runtime.h>

typedef __attribute__((ext_vector_type(16))) _Float16 v16h;
typedef __attribute__((ext_vector_type(8)))  float    v8f;

#define EPSV 1e-5f

__device__ __forceinline__ v8f wmma_f16(v16h a, v16h b, v8f c) {
    return __builtin_amdgcn_wmma_f32_16x16x32_f16(false, a, false, b, (short)0, c, false, false);
}

// CDNA5 16-bit A/B fragment swizzle: element e of lane (half = lane>>4)
// holds K = (e&7) | ((e&8)<<1) | (half<<3).  Gather 8 aligned dword pairs.
__device__ __forceinline__ v16h gather16(const _Float16* base, int half16) {
    union { unsigned int u[8]; v16h h; } r;
#pragma unroll
    for (int j = 0; j < 8; ++j) {
        int k = 2 * j + (half16 << 3) + ((j >= 4) ? 8 : 0);
        r.u[j] = *(const unsigned int*)(base + k);
    }
    return r.h;
}

// ---------------------------------------------------------------------------
// Prep: fold BN into per-channel alpha/beta:  y = relu(conv*alpha + beta)
// ---------------------------------------------------------------------------
__global__ void k_prep_ab(const float* c1b, const float* g1, const float* bb1, const float* m1, const float* v1,
                          const float* c2b, const float* g2, const float* bb2, const float* m2, const float* v2,
                          const float* c3b, const float* g3, const float* bb3, const float* m3, const float* v3,
                          float* ab1, float* ab2, float* ab3) {
    int t = threadIdx.x;
    if (t < 64) {
        float al = g1[t] * rsqrtf(v1[t] + EPSV);
        ab1[t] = al; ab1[64 + t] = (c1b[t] - m1[t]) * al + bb1[t];
    } else if (t < 192) {
        int c = t - 64;
        float al = g2[c] * rsqrtf(v2[c] + EPSV);
        ab2[c] = al; ab2[128 + c] = (c2b[c] - m2[c]) * al + bb2[c];
    } else if (t < 448) {
        int c = t - 192;
        float al = g3[c] * rsqrtf(v3[c] + EPSV);
        ab3[c] = al; ab3[256 + c] = (c3b[c] - m3[c]) * al + bb3[c];
    }
}

// ---------------------------------------------------------------------------
// Prep: rewrite weights into per-lane WMMA B-fragment layout (f16).
// frag element index t = ((kk*NT + nt)*32 + lane)*16 + e;
// value = W[k = kk*32 + swz(lane,e)][n = nt*16 + (lane&15)]
// ---------------------------------------------------------------------------
__device__ __forceinline__ void frag_decode(int t, int NT, int& k, int& n) {
    int e = t & 15, lane = (t >> 4) & 31, rest = t >> 9;
    int nt = rest % NT, kk = rest / NT;
    int half = lane >> 4;
    k = (kk << 5) + ((e & 7) | ((e & 8) << 1) | (half << 3));
    n = (nt << 4) + (lane & 15);
}

__global__ void k_prep_w1(const float* w, _Float16* out) {   // (64,1,1,51): K=64 pad, NT=4
    int t = blockIdx.x * 256 + threadIdx.x;
    if (t >= 2 * 4 * 512) return;
    int k, n; frag_decode(t, 4, k, n);
    out[t] = (_Float16)((k < 51) ? w[n * 51 + k] : 0.f);
}
__global__ void k_prep_w2(const float* w, _Float16* out) {   // (128,64,21): K=1344, NT=8
    int t = blockIdx.x * 256 + threadIdx.x;
    if (t >= 42 * 8 * 512) return;
    int k, n; frag_decode(t, 8, k, n);
    int tap = k >> 6, cin = k & 63;
    out[t] = (_Float16)w[(n * 64 + cin) * 21 + tap];
}
__global__ void k_prep_w3(const float* w, _Float16* out) {   // (256,128,9): K=1152, NT=16
    int t = blockIdx.x * 256 + threadIdx.x;
    if (t >= 36 * 16 * 512) return;
    int k, n; frag_decode(t, 16, k, n);
    int tap = k >> 7, cin = k & 127;
    out[t] = (_Float16)w[(n * 128 + cin) * 9 + tap];
}
__global__ void k_prep_win(const float* w, _Float16* out) {  // (1160,256): K=256, NT=73 (N pad 1168)
    int t = blockIdx.x * 256 + threadIdx.x;
    if (t >= 8 * 73 * 512) return;
    int k, n; frag_decode(t, 73, k, n);
    out[t] = (_Float16)((n < 1160) ? w[n * 256 + k] : 0.f);
}

// ---------------------------------------------------------------------------
// Kernel 1: conv1(1->64,k=51,s=4,p=25) + BN + ReLU + maxpool4 -> (B,2,240,64) f16
// GEMM: M=16 positions/wave, N=64 (4 ntiles), K=64 (51 zero-padded, 2 ksteps)
// Block = 4 waves = 64 pre-pool positions = 16 pooled positions.
// ---------------------------------------------------------------------------
__global__ void k_conv1(const float* __restrict__ x, const _Float16* __restrict__ w1f,
                        const float* __restrict__ ab1, _Float16* __restrict__ y1) {
    const int b = blockIdx.z, hrow = blockIdx.y, tile = blockIdx.x;   // tile 0..14
    const int w0 = tile * 64;
    __shared__ __align__(16) _Float16 xh[320];
    const float* xrow = x + (size_t)(b * 2 + hrow) * 3840;
    const int base = 4 * w0 - 25;
    for (int i = threadIdx.x; i < 320; i += 128) {
        int g = base + i;
        xh[i] = (g >= 0 && g < 3840) ? (_Float16)xrow[g] : (_Float16)0.f;
    }
    __syncthreads();

    const int lane = threadIdx.x & 31, wv = threadIdx.x >> 5;
    const int half = lane >> 4, r = lane & 15, mh = half * 8;
    const int pos = wv * 16 + r;                       // pre-pool position (A row)
    v16h a0 = gather16(xh + 4 * pos,      half);
    v16h a1 = gather16(xh + 4 * pos + 32, half);

    _Float16* dst = y1 + (size_t)(b * 2 + hrow) * 240 * 64;
    const int wpbase = tile * 16 + wv * 4 + (mh >> 2);
#pragma unroll
    for (int nt = 0; nt < 4; ++nt) {
        v8f c = {};
        v16h b0 = *(const v16h*)(w1f + ((size_t)(0 * 4 + nt) * 512 + lane * 16));
        v16h b1 = *(const v16h*)(w1f + ((size_t)(1 * 4 + nt) * 512 + lane * 16));
        c = wmma_f16(a0, b0, c);
        c = wmma_f16(a1, b1, c);
        int n = nt * 16 + r;
        float al = ab1[n], be = ab1[64 + n];
        float p0 = 0.f, p1 = 0.f;
#pragma unroll
        for (int e = 0; e < 8; ++e) {
            float v = fmaxf(c[e] * al + be, 0.f);
            if (e < 4) p0 = fmaxf(p0, v); else p1 = fmaxf(p1, v);
        }
        dst[(size_t)(wpbase + 0) * 64 + n] = (_Float16)p0;
        dst[(size_t)(wpbase + 1) * 64 + n] = (_Float16)p1;
    }
}

// ---------------------------------------------------------------------------
// Kernel 2: conv2(64->128,k=21,s=2,p=10) + BN + ReLU -> (B,2,120,128) f16
// Block per (b,h): im2col row staged in LDS (280x64 f16 incl. halos);
// 8 waves, wave wv owns N-tile wv; 42 ksteps x 8 M-tiles of WMMA.
// ---------------------------------------------------------------------------
__global__ void k_conv2(const _Float16* __restrict__ y1, const _Float16* __restrict__ w2f,
                        const float* __restrict__ ab2, _Float16* __restrict__ y2) {
    const int b = blockIdx.y, hrow = blockIdx.x;
    __shared__ __align__(16) _Float16 xs[280 * 64];
    uint4* xs4 = (uint4*)xs;
    const uint4* src4 = (const uint4*)(y1 + (size_t)(b * 2 + hrow) * 240 * 64);
    for (int i = threadIdx.x; i < 280 * 16; i += 256) {
        int rr = i >> 4, q = i & 15;
        int w = rr - 10;
        uint4 v = make_uint4(0, 0, 0, 0);
        if (w >= 0 && w < 240) v = src4[w * 16 + q];
        xs4[i] = v;
    }
    __syncthreads();

    const int lane = threadIdx.x & 31, wv = threadIdx.x >> 5;
    const int half = lane >> 4, r = lane & 15, mh = half * 8;
    v8f acc[8];
#pragma unroll
    for (int i = 0; i < 8; ++i) acc[i] = (v8f){};

    for (int kk = 0; kk < 42; ++kk) {
        int tap = kk >> 1, c0 = (kk & 1) << 5;
        v16h bf = *(const v16h*)(w2f + ((size_t)(kk * 8 + wv) * 512 + lane * 16));
#pragma unroll
        for (int mt = 0; mt < 8; ++mt) {
            int m = mt * 16 + r;                       // output position (A row)
            v16h a = gather16(xs + (2 * m + tap) * 64 + c0, half);
            acc[mt] = wmma_f16(a, bf, acc[mt]);
        }
    }

    const int n = wv * 16 + r;
    const float al = ab2[n], be = ab2[128 + n];
    _Float16* dst = y2 + (size_t)(b * 2 + hrow) * 120 * 128;
#pragma unroll
    for (int mt = 0; mt < 8; ++mt) {
#pragma unroll
        for (int e = 0; e < 8; ++e) {
            int m = mt * 16 + mh + e;
            if (m < 120) {
                float v = fmaxf(acc[mt][e] * al + be, 0.f);
                dst[(size_t)m * 128 + n] = (_Float16)v;
            }
        }
    }
}

// ---------------------------------------------------------------------------
// Kernel 3: conv3(128->256,k=9,s=2,p=4) + BN + ReLU + global mean -> (B,256) f32
// Block per b: both h rows (120 valid positions, 8 M-tiles), 16 ntiles, 2/wave.
// Mean reduction via LDS ds_add_f32.
// ---------------------------------------------------------------------------
__global__ void k_conv3(const _Float16* __restrict__ y2, const _Float16* __restrict__ w3f,
                        const float* __restrict__ ab3, float* __restrict__ hfeat) {
    const int b = blockIdx.x;
    __shared__ __align__(16) _Float16 xs[2 * 136 * 128];
    __shared__ float ssum[256];
    uint4* xs4 = (uint4*)xs;
    for (int i = threadIdx.x; i < 2 * 136 * 32; i += 256) {
        int hh = i / (136 * 32);
        int rem = i - hh * 136 * 32;
        int rr = rem >> 5, q = rem & 31;
        int w = rr - 4;
        uint4 v = make_uint4(0, 0, 0, 0);
        if (w >= 0 && w < 120)
            v = ((const uint4*)(y2 + (size_t)((b * 2 + hh) * 120 + w) * 128))[q];
        xs4[i] = v;
    }
    ssum[threadIdx.x] = 0.f;
    __syncthreads();

    const int lane = threadIdx.x & 31, wv = threadIdx.x >> 5;
    const int half = lane >> 4, r = lane & 15, mh = half * 8;
    v8f acc[2][8];
#pragma unroll
    for (int g = 0; g < 2; ++g)
#pragma unroll
        for (int i = 0; i < 8; ++i) acc[g][i] = (v8f){};

    int rowoff[8];
#pragma unroll
    for (int mt = 0; mt < 8; ++mt) {
        int p = mt * 16 + r;                           // global row 0..127 (valid <120)
        int hh = (p >= 60) ? 1 : 0;
        int m = p - 60 * hh; if (m > 59) m = 59;       // clamp masked rows
        rowoff[mt] = hh * (136 * 128) + 2 * m * 128;
    }

    for (int kk = 0; kk < 36; ++kk) {
        int tap = kk >> 2, c0 = (kk & 3) << 5;
        v16h bf0 = *(const v16h*)(w3f + ((size_t)(kk * 16 + wv)     * 512 + lane * 16));
        v16h bf1 = *(const v16h*)(w3f + ((size_t)(kk * 16 + wv + 8) * 512 + lane * 16));
#pragma unroll
        for (int mt = 0; mt < 8; ++mt) {
            v16h a = gather16(xs + rowoff[mt] + tap * 128 + c0, half);
            acc[0][mt] = wmma_f16(a, bf0, acc[0][mt]);
            acc[1][mt] = wmma_f16(a, bf1, acc[1][mt]);
        }
    }

#pragma unroll
    for (int g = 0; g < 2; ++g) {
        int n = (wv + 8 * g) * 16 + r;
        float al = ab3[n], be = ab3[256 + n];
        float s = 0.f;
#pragma unroll
        for (int mt = 0; mt < 8; ++mt) {
#pragma unroll
            for (int e = 0; e < 8; ++e) {
                int p = mt * 16 + mh + e;
                float v = fmaxf(acc[g][mt][e] * al + be, 0.f);
                if (p < 120) s += v;
            }
        }
        atomicAdd(&ssum[n], s);
    }
    __syncthreads();
    hfeat[(size_t)b * 256 + threadIdx.x] = ssum[threadIdx.x] * (1.f / 120.f);
}

// ---------------------------------------------------------------------------
// Kernel 4a: in_proj GEMM  (B=256 rows x N=1160 x K=256) with WMMA -> zxb f32
// Block of 16 batch rows; 8 ksteps; ntiles 0..72 split across 4 waves.
// ---------------------------------------------------------------------------
__global__ void k_inproj(const float* __restrict__ hfeat, const _Float16* __restrict__ winf,
                         float* __restrict__ zxb) {
    const int b0 = blockIdx.x * 16;
    __shared__ __align__(16) _Float16 hl[16 * 256];
    for (int i = threadIdx.x; i < 4096; i += 128)
        hl[i] = (_Float16)hfeat[(size_t)b0 * 256 + i];
    __syncthreads();

    const int lane = threadIdx.x & 31, wv = threadIdx.x >> 5;
    const int half = lane >> 4, r = lane & 15, mh = half * 8;
    v16h a[8];
#pragma unroll
    for (int kk = 0; kk < 8; ++kk)
        a[kk] = gather16(hl + r * 256 + kk * 32, half);

    for (int nt = wv; nt < 73; nt += 4) {
        v8f c = {};
#pragma unroll
        for (int kk = 0; kk < 8; ++kk) {
            v16h bf = *(const v16h*)(winf + ((size_t)(kk * 73 + nt) * 512 + lane * 16));
            c = wmma_f16(a[kk], bf, c);
        }
        int j = nt * 16 + r;
        if (j < 1160) {
#pragma unroll
            for (int e = 0; e < 8; ++e)
                zxb[(size_t)(b0 + mh + e) * 1160 + j] = c[e];
        }
    }
}

// ---------------------------------------------------------------------------
// Kernel 4b: Mamba2 (L=1) elementwise + RMSNorm + out_proj + MLP head -> (B,1)
// L=1 => state starts at 0 => y = dt*x*(B.C) + D*x  (dA unused).
// One 256-thread block per batch.
// ---------------------------------------------------------------------------
__global__ void k_mamba_head(const float* __restrict__ zxb,
                             const float* __restrict__ mconv_w, const float* __restrict__ mconv_b,
                             const float* __restrict__ mdt_bias, const float* __restrict__ mD,
                             const float* __restrict__ mnorm_w, const float* __restrict__ mw_out,
                             const float* __restrict__ f1w, const float* __restrict__ f1b,
                             const float* __restrict__ bn4g, const float* __restrict__ bn4b,
                             const float* __restrict__ bn4m, const float* __restrict__ bn4v,
                             const float* __restrict__ f2w, const float* __restrict__ f2b,
                             float* __restrict__ out) {
    const int b = blockIdx.x;
    const float* row = zxb + (size_t)b * 1160;
    __shared__ float sxin[512], sB[64], sC[64], sdt[8], sy[512], so[256];
    __shared__ float sbc, sss, sfin;
    const int tid = threadIdx.x;
    if (tid == 0) { sbc = 0.f; sss = 0.f; sfin = 0.f; }

    // depthwise conv (only tap 3 sees data at t=0 with left zero-pad) + silu
    for (int j = tid; j < 640; j += 256) {
        float v = row[512 + j] * mconv_w[j * 4 + 3] + mconv_b[j];
        float s = v / (1.f + expf(-v));
        if (j < 512) sxin[j] = s;
        else if (j < 576) sB[j - 512] = s;
        else sC[j - 576] = s;
    }
    if (tid < 8) {
        float d = row[1152 + tid] + mdt_bias[tid];
        sdt[tid] = (d > 20.f) ? d : log1pf(expf(d));
    }
    __syncthreads();

    if (tid < 64) atomicAdd(&sbc, sB[tid] * sC[tid]);
    __syncthreads();
    const float bc = sbc;

    float lss = 0.f;
    for (int i = tid; i < 512; i += 256) {
        int hd = i >> 6;
        float xh = sxin[i];
        float yv = sdt[hd] * xh * bc + mD[hd] * xh;
        float z = row[i];
        yv *= z / (1.f + expf(-z));                    // gate: y * silu(z)
        sy[i] = yv;
        lss += yv * yv;
    }
    atomicAdd(&sss, lss);
    __syncthreads();
    const float scale = rsqrtf(sss * (1.f / 512.f) + EPSV);
    for (int i = tid; i < 512; i += 256) sy[i] = sy[i] * scale * mnorm_w[i];
    __syncthreads();

    {   // out_proj: (512 -> 256)
        float acc = 0.f;
        const float* wr = mw_out + (size_t)tid * 512;
        for (int k = 0; k < 512; ++k) acc += sy[k] * wr[k];
        so[tid] = acc;
    }
    __syncthreads();

    if (tid < 64) {                                    // fc1 + BN + ReLU + fc2 partial
        float acc = f1b[tid];
        const float* wr = f1w + tid * 256;
        for (int c = 0; c < 256; ++c) acc += so[c] * wr[c];
        float al = bn4g[tid] * rsqrtf(bn4v[tid] + EPSV);
        acc = fmaxf((acc - bn4m[tid]) * al + bn4b[tid], 0.f);
        atomicAdd(&sfin, acc * f2w[tid]);
    }
    __syncthreads();
    if (tid == 0) out[b] = sfin + f2b[0];
}

// ---------------------------------------------------------------------------
extern "C" void kernel_launch(void* const* d_in, const int* in_sizes, int n_in,
                              void* d_out, int out_size, void* d_ws, size_t ws_size,
                              hipStream_t stream) {
    const float* x    = (const float*)d_in[0];
    const float* c1w  = (const float*)d_in[1];
    const float* c1b  = (const float*)d_in[2];
    const float* bn1g = (const float*)d_in[3];
    const float* bn1b = (const float*)d_in[4];
    const float* bn1m = (const float*)d_in[5];
    const float* bn1v = (const float*)d_in[6];
    const float* c2w  = (const float*)d_in[7];
    const float* c2b  = (const float*)d_in[8];
    const float* bn2g = (const float*)d_in[9];
    const float* bn2b = (const float*)d_in[10];
    const float* bn2m = (const float*)d_in[11];
    const float* bn2v = (const float*)d_in[12];
    const float* c3w  = (const float*)d_in[13];
    const float* c3b  = (const float*)d_in[14];
    const float* bn3g = (const float*)d_in[15];
    const float* bn3b = (const float*)d_in[16];
    const float* bn3m = (const float*)d_in[17];
    const float* bn3v = (const float*)d_in[18];
    const float* mw_in   = (const float*)d_in[19];
    const float* mconv_w = (const float*)d_in[20];
    const float* mconv_b = (const float*)d_in[21];
    const float* mdt_b   = (const float*)d_in[22];
    /* mA_log d_in[23] unused: state starts at zero with L=1 */
    const float* mD      = (const float*)d_in[24];
    const float* mnorm_w = (const float*)d_in[25];
    const float* mw_out  = (const float*)d_in[26];
    const float* f1w  = (const float*)d_in[27];
    const float* f1b  = (const float*)d_in[28];
    const float* bn4g = (const float*)d_in[29];
    const float* bn4b = (const float*)d_in[30];
    const float* bn4m = (const float*)d_in[31];
    const float* bn4v = (const float*)d_in[32];
    const float* f2w  = (const float*)d_in[33];
    const float* f2b  = (const float*)d_in[34];
    float* out = (float*)d_out;

    // workspace carve-out (256B aligned)
    char* ws = (char*)d_ws;
    size_t off = 0;
    auto carve = [&](size_t bytes) {
        char* p = ws + off;
        off = (off + bytes + 255) & ~(size_t)255;
        return p;
    };
    _Float16* w1f  = (_Float16*)carve(2  * 4  * 512 * sizeof(_Float16));
    _Float16* w2f  = (_Float16*)carve(42 * 8  * 512 * sizeof(_Float16));
    _Float16* w3f  = (_Float16*)carve(36 * 16 * 512 * sizeof(_Float16));
    _Float16* winf = (_Float16*)carve(8  * 73 * 512 * sizeof(_Float16));
    float* ab1 = (float*)carve(128 * sizeof(float));
    float* ab2 = (float*)carve(256 * sizeof(float));
    float* ab3 = (float*)carve(512 * sizeof(float));
    _Float16* y1 = (_Float16*)carve((size_t)256 * 2 * 240 * 64  * sizeof(_Float16));
    _Float16* y2 = (_Float16*)carve((size_t)256 * 2 * 120 * 128 * sizeof(_Float16));
    float* hfeat = (float*)carve((size_t)256 * 256  * sizeof(float));
    float* zxb   = (float*)carve((size_t)256 * 1160 * sizeof(float));
    (void)ws_size; (void)in_sizes; (void)n_in; (void)out_size;

    // weight/BN prep
    k_prep_ab<<<1, 448, 0, stream>>>(c1b, bn1g, bn1b, bn1m, bn1v,
                                     c2b, bn2g, bn2b, bn2m, bn2v,
                                     c3b, bn3g, bn3b, bn3m, bn3v, ab1, ab2, ab3);
    k_prep_w1 <<<(2 * 4 * 512 + 255) / 256,  256, 0, stream>>>(c1w,  w1f);
    k_prep_w2 <<<(42 * 8 * 512) / 256,       256, 0, stream>>>(c2w,  w2f);
    k_prep_w3 <<<(36 * 16 * 512) / 256,      256, 0, stream>>>(c3w,  w3f);
    k_prep_win<<<(8 * 73 * 512) / 256,       256, 0, stream>>>(mw_in, winf);

    // pipeline
    k_conv1<<<dim3(15, 2, 256), 128, 0, stream>>>(x, w1f, ab1, y1);
    k_conv2<<<dim3(2, 256),     256, 0, stream>>>(y1, w2f, ab2, y2);
    k_conv3<<<256,              256, 0, stream>>>(y2, w3f, ab3, hfeat);
    k_inproj<<<16,              128, 0, stream>>>(hfeat, winf, zxb);
    k_mamba_head<<<256, 256, 0, stream>>>(zxb, mconv_w, mconv_b, mdt_b, mD, mnorm_w,
                                          mw_out, f1w, f1b, bn4g, bn4b, bn4m, bn4v,
                                          f2w, f2b, out);
}
